// CrossAttention_86122684220038
// MI455X (gfx1250) — matrix operand order repaired and checked by reading, not compiled, not run on hardware
//
#include <hip/hip_runtime.h>

typedef __attribute__((ext_vector_type(4)))  __bf16 v4bf;
typedef __attribute__((ext_vector_type(8)))  __bf16 v8bf;
typedef __attribute__((ext_vector_type(16))) __bf16 v16bf;
typedef __attribute__((ext_vector_type(8)))  float  v8f;

#define BM    128
#define BN    64
#define BK    64
#define LDAF  72   // f32->bf16 staged A tile (manual, 8B aligned)
#define LDAP  80   // async-staged A tile (16B aligned rows)
#define LDBW  80   // async-staged B tile (16B aligned rows)
#define LDK   80   // async-staged K tile in attention
#define CSLD  68
#define LOG2E 1.4426950408889634f

// ---------------- helpers ----------------

__device__ __forceinline__ v16bf pack16(v4bf a, v4bf b, v4bf c, v4bf d) {
  v16bf f;
#pragma unroll
  for (int i = 0; i < 4; ++i) { f[i] = a[i]; f[i+4] = b[i]; f[i+8] = c[i]; f[i+12] = d[i]; }
  return f;
}

// A-fragment (16x32 bf16): lanes 0-15 hold K {0..7,16..23}, lanes 16-31 K {8..15,24..31}
__device__ __forceinline__ v16bf frag_a_row(const __bf16* prow, int hlf) {
  const __bf16* p = prow + hlf * 8;
  return pack16(*(const v4bf*)(p),      *(const v4bf*)(p + 4),
                *(const v4bf*)(p + 16), *(const v4bf*)(p + 20));
}

// B-fragment (32x16 bf16): lanes 0-15 hold K {0..7,16..23}, lanes 16-31 K {8..15,24..31}
__device__ __forceinline__ v16bf frag_b_row(const __bf16* prow, int hlf) {
  const __bf16* p = prow + hlf * 8;
  return pack16(*(const v4bf*)(p),      *(const v4bf*)(p + 4),
                *(const v4bf*)(p + 16), *(const v4bf*)(p + 20));
}

__device__ __forceinline__ v8f wmma_bf16(v16bf a, v16bf b, v8f c) {
  return __builtin_amdgcn_wmma_f32_16x16x32_bf16(false, a, false, b, (short)0, c, false, false);
}

__device__ __forceinline__ v4bf cvt4(float4 f) {
  v4bf r; r[0] = (__bf16)f.x; r[1] = (__bf16)f.y; r[2] = (__bf16)f.z; r[3] = (__bf16)f.w;
  return r;
}
__device__ __forceinline__ v4bf cvt4s(float a, float b, float c, float d) {
  v4bf r; r[0] = (__bf16)a; r[1] = (__bf16)b; r[2] = (__bf16)c; r[3] = (__bf16)d;
  return r;
}

template <int M>
__device__ __forceinline__ float swzx(float v) {
  return __int_as_float(__builtin_amdgcn_ds_swizzle(__float_as_int(v), (M << 10) | 0x1F));
}
__device__ __forceinline__ float red16max(float v) {
  v = fmaxf(v, swzx<1>(v)); v = fmaxf(v, swzx<2>(v));
  v = fmaxf(v, swzx<4>(v)); v = fmaxf(v, swzx<8>(v));
  return v;
}
__device__ __forceinline__ float red16add(float v) {
  v += swzx<1>(v); v += swzx<2>(v); v += swzx<4>(v); v += swzx<8>(v);
  return v;
}

// CDNA5 async global->LDS copy (16B per lane), tracked by ASYNCcnt.
// Generic shared pointer's low 32 bits are the wave-relative LDS offset (flat->LDS rule).
__device__ __forceinline__ void async_ld_b128(void* lds_dst, const void* gsrc) {
  unsigned loff = (unsigned)(unsigned long long)lds_dst;
  unsigned long long g = (unsigned long long)gsrc;
  asm volatile("global_load_async_to_lds_b128 %0, %1, off" :: "v"(loff), "v"(g) : "memory");
}
__device__ __forceinline__ void wait_async() {
  asm volatile("s_wait_asynccnt 0" ::: "memory");
}

// ---------------- weight transpose+convert: Wt[n][k] = (bf16)W[k][n] ----------------

__global__ __launch_bounds__(256) void k_wconv(const float* __restrict__ W,
                                               __bf16* __restrict__ Wt, int N, int K) {
  long idx = (long)blockIdx.x * 256 + threadIdx.x;
  int k = (int)(idx / N), n = (int)(idx % N);
  Wt[(size_t)n * K + k] = (__bf16)W[idx];
}

// ---------------- GEMM mainloop: C(128x64) tile into LDS Cs, f32 A input ----------------

__device__ __forceinline__ void gemm_tile_f32A(const float* __restrict__ Ag,
                                               const __bf16* __restrict__ Btg, int Kd,
                                               __bf16* Alds, __bf16* Btlds, float* Cs) {
  int tid = threadIdx.x;
  int lane = tid & 31, wv = tid >> 5;
  int hlf = lane >> 4, nl = lane & 15;
  int wm = wv & 3, wn = wv >> 2;

  v8f acc[2][2];
#pragma unroll
  for (int i = 0; i < 2; ++i)
#pragma unroll
    for (int j = 0; j < 2; ++j)
#pragma unroll
      for (int e = 0; e < 8; ++e) acc[i][j][e] = 0.f;

  int arow = tid >> 1, ahf = tid & 1;   // 128 rows x 32 floats
  int brow = tid >> 2, bq = tid & 3;    // 64 rows x 2 16B-chunks

  for (int k0 = 0; k0 < Kd; k0 += BK) {
    __syncthreads();
    {
      // B tile: pure bf16 copy -> async DMA to LDS
      const __bf16* bs = Btg + (size_t)brow * Kd + k0;
#pragma unroll
      for (int c = 0; c < 2; ++c) {
        int ck = (bq * 2 + c) * 8;
        async_ld_b128(Btlds + brow * LDBW + ck, bs + ck);
      }
      // A tile: f32 -> bf16 conversion, manual staging
      const float* src = Ag + (size_t)arow * Kd + k0 + ahf * 32;
      __bf16* dst = Alds + arow * LDAF + ahf * 32;
#pragma unroll
      for (int c = 0; c < 8; ++c) {
        float4 f = ((const float4*)src)[c];
        *(v4bf*)(dst + c * 4) = cvt4(f);
      }
      if (k0 + BK < Kd) __builtin_prefetch(src + BK, 0, 0);
    }
    wait_async();
    __syncthreads();
#pragma unroll
    for (int kc = 0; kc < 2; ++kc) {
      v16bf a0 = frag_a_row(Alds + (wm * 32 + nl) * LDAF + kc * 32, hlf);
      v16bf a1 = frag_a_row(Alds + (wm * 32 + 16 + nl) * LDAF + kc * 32, hlf);
      v16bf b0 = frag_b_row(Btlds + (wn * 32 + nl) * LDBW + kc * 32, hlf);
      v16bf b1 = frag_b_row(Btlds + (wn * 32 + 16 + nl) * LDBW + kc * 32, hlf);
      acc[0][0] = wmma_bf16(a0, b0, acc[0][0]);
      acc[0][1] = wmma_bf16(a0, b1, acc[0][1]);
      acc[1][0] = wmma_bf16(a1, b0, acc[1][0]);
      acc[1][1] = wmma_bf16(a1, b1, acc[1][1]);
    }
  }
  __syncthreads();
#pragma unroll
  for (int ti = 0; ti < 2; ++ti)
#pragma unroll
    for (int tj = 0; tj < 2; ++tj)
#pragma unroll
      for (int e = 0; e < 8; ++e)
        Cs[(wm * 32 + ti * 16 + e + 8 * hlf) * CSLD + wn * 32 + tj * 16 + nl] = acc[ti][tj][e];
  __syncthreads();
}

// ---------------- Q projection + per-head RMSNorm ----------------

__global__ __launch_bounds__(256) void k_gemm_q(const float* __restrict__ X,
                                                const __bf16* __restrict__ WqT,
                                                const float* __restrict__ qgamma,
                                                __bf16* __restrict__ Qo) {
  __shared__ __bf16 Alds[BM * LDAF];
  __shared__ __bf16 Btlds[BN * LDBW];
  __shared__ float  Cs[BM * CSLD];
  int mbase = blockIdx.x * BM;
  int h = blockIdx.y;  // one head per 64-col tile
  gemm_tile_f32A(X + (size_t)mbase * 1024, WqT + (size_t)h * 64 * 1024, 1024, Alds, Btlds, Cs);

  int tid = threadIdx.x;
  int rr = tid >> 1, hf = tid & 1;
  const float* crow = Cs + rr * CSLD + hf * 32;
  float vals[32]; float ss = 0.f;
#pragma unroll
  for (int j = 0; j < 32; ++j) { float t = crow[j]; vals[j] = t; ss += t * t; }
  ss += swzx<1>(ss);                       // combine with other half of the row
  float inv = rsqrtf(ss * (1.f / 64.f) + 1e-6f);
  int r = mbase + rr;
  int b = r >> 12, l = r & 4095;           // L = 4096
  __bf16* dst = Qo + (((size_t)(b * 16 + h) * 4096 + l) * 64 + hf * 32);
#pragma unroll
  for (int j = 0; j < 8; ++j) {
    const float* v4 = vals + j * 4;
    const float* g4 = qgamma + hf * 32 + j * 4;
    *(v4bf*)(dst + j * 4) = cvt4s(v4[0]*inv*g4[0], v4[1]*inv*g4[1],
                                  v4[2]*inv*g4[2], v4[3]*inv*g4[3]);
  }
}

// ---------------- KV projection (+ RMSNorm for K) ----------------

__global__ __launch_bounds__(256) void k_gemm_kv(const float* __restrict__ C,
                                                 const __bf16* __restrict__ WkvT,
                                                 const float* __restrict__ kgamma,
                                                 __bf16* __restrict__ Ko,
                                                 __bf16* __restrict__ Vo) {
  __shared__ __bf16 Alds[BM * LDAF];
  __shared__ __bf16 Btlds[BN * LDBW];
  __shared__ float  Cs[BM * CSLD];
  int mbase = blockIdx.x * BM;
  int nt = blockIdx.y;  // 0..15 -> K head nt ; 16..31 -> V head nt-16
  gemm_tile_f32A(C + (size_t)mbase * 1024, WkvT + (size_t)nt * 64 * 1024, 1024, Alds, Btlds, Cs);

  int tid = threadIdx.x;
  int rr = tid >> 1, hf = tid & 1;
  const float* crow = Cs + rr * CSLD + hf * 32;
  float vals[32]; float ss = 0.f;
#pragma unroll
  for (int j = 0; j < 32; ++j) { float t = crow[j]; vals[j] = t; ss += t * t; }
  ss += swzx<1>(ss);
  int r = mbase + rr;
  int b = r >> 10, m = r & 1023;           // Lc = 1024
  if (nt < 16) {
    float inv = rsqrtf(ss * (1.f / 64.f) + 1e-6f);
    __bf16* dst = Ko + (((size_t)(b * 16 + nt) * 1024 + m) * 64 + hf * 32);
#pragma unroll
    for (int j = 0; j < 8; ++j) {
      const float* v4 = vals + j * 4;
      const float* g4 = kgamma + hf * 32 + j * 4;
      *(v4bf*)(dst + j * 4) = cvt4s(v4[0]*inv*g4[0], v4[1]*inv*g4[1],
                                    v4[2]*inv*g4[2], v4[3]*inv*g4[3]);
    }
  } else {
    int h = nt - 16;
    __bf16* dst = Vo + (((size_t)(b * 16 + h) * 1024 + m) * 64 + hf * 32);
#pragma unroll
    for (int j = 0; j < 8; ++j) {
      const float* v4 = vals + j * 4;
      *(v4bf*)(dst + j * 4) = cvt4s(v4[0], v4[1], v4[2], v4[3]);
    }
  }
}

// ---------------- flash attention: softmax(QK^T/8) V ----------------

__global__ __launch_bounds__(128) void k_attn(const __bf16* __restrict__ Q,
                                              const __bf16* __restrict__ Kb,
                                              const __bf16* __restrict__ Vb,
                                              __bf16* __restrict__ OB) {
  __shared__ __bf16 Klds[32 * LDK];      // [m][d], async-staged
  __shared__ __bf16 Vtlds[64 * 40];      // [d][m], transposed during staging
  __shared__ __bf16 Plds[4 * 16 * 40];   // per-wave [row][m]

  int tid = threadIdx.x;
  int lane = tid & 31, wv = tid >> 5;
  int hlf = lane >> 4, nl = lane & 15;
  int bh = blockIdx.y;
  int lbase = blockIdx.x * 64;

  const __bf16* qg = Q + ((size_t)bh * 4096 + lbase) * 64;
  const __bf16* kg = Kb + (size_t)bh * 1024 * 64;
  const __bf16* vg = Vb + (size_t)bh * 1024 * 64;

  // hoist Q fragments: rows wv*16 + nl, d-chunks 0..31 / 32..63
  const __bf16* qrow = qg + (size_t)(wv * 16 + nl) * 64;
  v16bf qa0 = frag_a_row(qrow, hlf);
  v16bf qa1 = frag_a_row(qrow + 32, hlf);

  v8f o[4];
  float rm[8], rl[8];
#pragma unroll
  for (int dt = 0; dt < 4; ++dt)
#pragma unroll
    for (int e = 0; e < 8; ++e) o[dt][e] = 0.f;
#pragma unroll
  for (int e = 0; e < 8; ++e) { rm[e] = -1e30f; rl[e] = 0.f; }

  const float scale = 0.125f;               // 1/sqrt(64)
  int sm = tid >> 2, sq = tid & 3;          // staging: 32 rows, 4 threads/row

  for (int mb = 0; mb < 1024; mb += 32) {
    __syncthreads();
    {
      // K chunk: direct bf16 copy -> async DMA (2 x 16B per thread)
      const __bf16* ks = kg + ((size_t)(mb + sm)) * 64;
#pragma unroll
      for (int c = 0; c < 2; ++c) {
        int ck = (sq * 2 + c) * 8;
        async_ld_b128(Klds + sm * LDK + ck, ks + ck);
      }
      // V chunk: transpose into LDS (manual scatter)
      int sd = sq * 16;
      const __bf16* vs = vg + ((size_t)(mb + sm)) * 64 + sd;
      v4bf w0 = *(const v4bf*)(vs), w1 = *(const v4bf*)(vs + 4);
      v4bf w2 = *(const v4bf*)(vs + 8), w3 = *(const v4bf*)(vs + 12);
#pragma unroll
      for (int j = 0; j < 4; ++j) {
        Vtlds[(sd + j) * 40 + sm]      = w0[j];
        Vtlds[(sd + 4 + j) * 40 + sm]  = w1[j];
        Vtlds[(sd + 8 + j) * 40 + sm]  = w2[j];
        Vtlds[(sd + 12 + j) * 40 + sm] = w3[j];
      }
      if (mb + 32 < 1024) __builtin_prefetch(vs + 32 * 64, 0, 0);
    }
    wait_async();
    __syncthreads();

    // S = Q K^T for two 16-wide m sub-tiles
    v8f s0, s1;
#pragma unroll
    for (int e = 0; e < 8; ++e) { s0[e] = 0.f; s1[e] = 0.f; }
    {
      v16bf kb0 = frag_b_row(Klds + (nl) * LDK, hlf);
      v16bf kb1 = frag_b_row(Klds + (16 + nl) * LDK, hlf);
      s0 = wmma_bf16(qa0, kb0, s0);
      s1 = wmma_bf16(qa0, kb1, s1);
      kb0 = frag_b_row(Klds + (nl) * LDK + 32, hlf);
      kb1 = frag_b_row(Klds + (16 + nl) * LDK + 32, hlf);
      s0 = wmma_bf16(qa1, kb0, s0);
      s1 = wmma_bf16(qa1, kb1, s1);
    }

    // online softmax (one row per C-VGPR, rows live across 16-lane groups)
    float alpha[8];
    __bf16* pw = Plds + wv * 16 * 40;
#pragma unroll
    for (int e = 0; e < 8; ++e) {
      float a = s0[e] * scale, b = s1[e] * scale;
      float mx = red16max(fmaxf(a, b));
      float nm = fmaxf(rm[e], mx);
      alpha[e] = exp2f((rm[e] - nm) * LOG2E);
      rm[e] = nm;
      float p0 = exp2f((a - nm) * LOG2E);
      float p1 = exp2f((b - nm) * LOG2E);
      float sum = red16add(p0 + p1);
      rl[e] = rl[e] * alpha[e] + sum;
      int row = e + 8 * hlf;
      pw[row * 40 + nl] = (__bf16)p0;
      pw[row * 40 + 16 + nl] = (__bf16)p1;
    }
#pragma unroll
    for (int dt = 0; dt < 4; ++dt)
#pragma unroll
      for (int e = 0; e < 8; ++e) o[dt][e] *= alpha[e];

    __syncthreads();

    // O += P * V
    v16bf pf = frag_a_row(pw + nl * 40, hlf);
#pragma unroll
    for (int dt = 0; dt < 4; ++dt) {
      v16bf vf = frag_b_row(Vtlds + (dt * 16 + nl) * 40, hlf);
      o[dt] = wmma_bf16(pf, vf, o[dt]);
    }
  }

  int b = bh >> 4, h = bh & 15;
#pragma unroll
  for (int e = 0; e < 8; ++e) {
    float inv = 1.f / rl[e];
    int l = lbase + wv * 16 + e + 8 * hlf;
    __bf16* dst = OB + ((size_t)(b * 4096 + l)) * 1024 + h * 64 + nl;
#pragma unroll
    for (int dt = 0; dt < 4; ++dt) dst[dt * 16] = (__bf16)(o[dt][e] * inv);
  }
}

// ---------------- output projection (bf16 A), direct f32 store ----------------

__global__ __launch_bounds__(256) void k_gemm_proj(const __bf16* __restrict__ A,
                                                   const __bf16* __restrict__ WpT,
                                                   float* __restrict__ Out) {
  __shared__ __bf16 Alds[BM * LDAP];
  __shared__ __bf16 Btlds[BN * LDBW];
  int tid = threadIdx.x;
  int lane = tid & 31, wv = tid >> 5;
  int hlf = lane >> 4, nl = lane & 15;
  int wm = wv & 3, wn = wv >> 2;
  int mbase = blockIdx.x * BM, nbase = blockIdx.y * BN;

  const __bf16* Ag = A + (size_t)mbase * 1024;
  const __bf16* Btg = WpT + (size_t)nbase * 1024;

  v8f acc[2][2];
#pragma unroll
  for (int i = 0; i < 2; ++i)
#pragma unroll
    for (int j = 0; j < 2; ++j)
#pragma unroll
      for (int e = 0; e < 8; ++e) acc[i][j][e] = 0.f;

  int arow = tid >> 1, aq = tid & 1;   // 128 rows, 2 threads/row, 4 x 16B each
  int brow = tid >> 2, bq = tid & 3;   // 64 rows, 4 threads/row, 2 x 16B each

  for (int k0 = 0; k0 < 1024; k0 += BK) {
    __syncthreads();
    {
      const __bf16* as = Ag + (size_t)arow * 1024 + k0;
#pragma unroll
      for (int c = 0; c < 4; ++c) {
        int ck = (aq * 4 + c) * 8;
        async_ld_b128(Alds + arow * LDAP + ck, as + ck);
      }
      const __bf16* bs = Btg + (size_t)brow * 1024 + k0;
#pragma unroll
      for (int c = 0; c < 2; ++c) {
        int ck = (bq * 2 + c) * 8;
        async_ld_b128(Btlds + brow * LDBW + ck, bs + ck);
      }
    }
    wait_async();
    __syncthreads();
#pragma unroll
    for (int kc = 0; kc < 2; ++kc) {
      v16bf a0 = frag_a_row(Alds + (wm * 32 + nl) * LDAP + kc * 32, hlf);
      v16bf a1 = frag_a_row(Alds + (wm * 32 + 16 + nl) * LDAP + kc * 32, hlf);
      v16bf b0 = frag_b_row(Btlds + (wn * 32 + nl) * LDBW + kc * 32, hlf);
      v16bf b1 = frag_b_row(Btlds + (wn * 32 + 16 + nl) * LDBW + kc * 32, hlf);
      acc[0][0] = wmma_bf16(a0, b0, acc[0][0]);
      acc[0][1] = wmma_bf16(a0, b1, acc[0][1]);
      acc[1][0] = wmma_bf16(a1, b0, acc[1][0]);
      acc[1][1] = wmma_bf16(a1, b1, acc[1][1]);
    }
  }
#pragma unroll
  for (int ti = 0; ti < 2; ++ti)
#pragma unroll
    for (int tj = 0; tj < 2; ++tj)
#pragma unroll
      for (int e = 0; e < 8; ++e) {
        int r = mbase + wm * 32 + ti * 16 + e + 8 * hlf;
        int c = nbase + wn * 32 + tj * 16 + nl;
        Out[(size_t)r * 1024 + c] = acc[ti][tj][e];
      }
}

// ---------------- host launcher ----------------

extern "C" void kernel_launch(void* const* d_in, const int* in_sizes, int n_in,
                              void* d_out, int out_size, void* d_ws, size_t ws_size,
                              hipStream_t stream) {
  const float* x      = (const float*)d_in[0];
  const float* ctx    = (const float*)d_in[1];
  const float* Wq     = (const float*)d_in[2];
  const float* Wkv    = (const float*)d_in[3];
  const float* Wproj  = (const float*)d_in[4];
  const float* qgamma = (const float*)d_in[5];
  const float* kgamma = (const float*)d_in[6];
  float* out = (float*)d_out;

  char* ws = (char*)d_ws;
  __bf16* WqT    = (__bf16*)(ws);                          // 2 MiB
  __bf16* WkvT   = (__bf16*)(ws + (size_t)2  * (1 << 20)); // 4 MiB
  __bf16* WprojT = (__bf16*)(ws + (size_t)6  * (1 << 20)); // 2 MiB
  __bf16* Qb     = (__bf16*)(ws + (size_t)8  * (1 << 20)); // 32 MiB
  __bf16* Kb     = (__bf16*)(ws + (size_t)40 * (1 << 20)); // 8 MiB
  __bf16* Vb     = (__bf16*)(ws + (size_t)48 * (1 << 20)); // 8 MiB
  __bf16* OB     = (__bf16*)(ws + (size_t)56 * (1 << 20)); // 32 MiB

  k_wconv<<<4096, 256, 0, stream>>>(Wq, WqT, 1024, 1024);
  k_wconv<<<8192, 256, 0, stream>>>(Wkv, WkvT, 2048, 1024);
  k_wconv<<<4096, 256, 0, stream>>>(Wproj, WprojT, 1024, 1024);

  k_gemm_q<<<dim3(128, 16), 256, 0, stream>>>(x, WqT, qgamma, Qb);
  k_gemm_kv<<<dim3(32, 32), 256, 0, stream>>>(ctx, WkvT, kgamma, Kb, Vb);
  k_attn<<<dim3(64, 64), 128, 0, stream>>>(Qb, Kb, Vb, OB);
  k_gemm_proj<<<dim3(128, 16), 256, 0, stream>>>(OB, WprojT, out);
}
